// ParallelSlowCastle_18622978196194
// MI455X (gfx1250) — matrix operand order repaired and checked by reading, not compiled, not run on hardware
//
#include <hip/hip_runtime.h>
#include <cmath>

typedef __attribute__((ext_vector_type(16))) __bf16 v16bf;
typedef __attribute__((ext_vector_type(8)))  float  v8f;
typedef __attribute__((ext_vector_type(8)))  __bf16 bf16x8;
typedef int v4i __attribute__((vector_size(16)));  // matches async-LDS builtin param

#define WMMA_BF16(a, b, c) \
  __builtin_amdgcn_wmma_f32_16x16x32_bf16(false, (a), false, (b), (short)0, (c), false, false)

// ---------------------------------------------------------------------------
// CDNA5 async global->LDS path (ASYNCcnt). Falls back to synchronous b128
// copies if the builtin is unavailable (e.g. host parse pass).
// ---------------------------------------------------------------------------
#if defined(__HIP_DEVICE_COMPILE__) && \
    __has_builtin(__builtin_amdgcn_global_load_async_to_lds_b128)
#define HAVE_ASYNC_LDS 1
#else
#define HAVE_ASYNC_LDS 0
#endif

__device__ __forceinline__ void async_b128(void* lds, const void* g) {
#if HAVE_ASYNC_LDS
  __builtin_amdgcn_global_load_async_to_lds_b128(
      (__attribute__((address_space(1))) v4i*)(void*)g,
      (__attribute__((address_space(3))) v4i*)lds,
      0, 0);
#else
  *(bf16x8*)lds = *(const bf16x8*)g;
#endif
}

__device__ __forceinline__ void wait_async0() {
#if HAVE_ASYNC_LDS
#if __has_builtin(__builtin_amdgcn_s_wait_asynccnt)
  __builtin_amdgcn_s_wait_asynccnt(0);
#else
  asm volatile("s_wait_asynccnt 0x0" ::: "memory");
#endif
#endif
}

__device__ __forceinline__ int lane_id() { return (int)(threadIdx.x & 31u); }

__device__ __forceinline__ v8f zero_v8f() {
  v8f z;
#pragma unroll
  for (int e = 0; e < 8; ++e) z[e] = 0.f;
  return z;
}

// A fragment: 16(M) x 32(K) bf16, source row-major with leading dim ld.
// Lane m (0..15): K = 0..15; lanes 16..31: K = 16..31 (contiguous).
__device__ __forceinline__ v16bf load_A_frag(const __bf16* __restrict__ p, int ld) {
  const int l = lane_id();
  const int m = l & 15;
  const int kh = (l >> 4) * 16;
  const __bf16* row = p + (size_t)m * ld;
  v16bf a;
#pragma unroll
  for (int v = 0; v < 8; ++v) {
    const int kb = kh + v * 2;
    a[2 * v]     = row[kb];
    a[2 * v + 1] = row[kb + 1];
  }
  return a;
}

// B fragment 32(K) x 16(N) where the SOURCE is stored [N][K] row-major
// (multiply by the transpose of the stored matrix). Contiguous per-lane reads.
__device__ __forceinline__ v16bf load_B_frag_T(const __bf16* __restrict__ p, int ld) {
  const int l = lane_id();
  const int n = l & 15;
  const int kh = (l >> 4) * 16;
  const __bf16* row = p + (size_t)n * ld + kh;
  v16bf b;
#pragma unroll
  for (int v = 0; v < 16; ++v) b[v] = row[v];
  return b;
}

// B fragment 32(K) x 16(N) where the SOURCE is stored [K][N] row-major.
__device__ __forceinline__ v16bf load_B_frag_N(const __bf16* __restrict__ p, int ld) {
  const int l = lane_id();
  const int n = l & 15;
  const int kh = (l >> 4) * 16;
  v16bf b;
#pragma unroll
  for (int v = 0; v < 16; ++v) b[v] = p[(size_t)(kh + v) * ld + n];
  return b;
}

// ---------------------------------------------------------------------------
// Elementwise conversion kernels
// ---------------------------------------------------------------------------
__global__ __launch_bounds__(256) void f32_to_bf16_kernel(const float* __restrict__ in,
                                                          __bf16* __restrict__ out,
                                                          size_t n) {
  size_t i = (size_t)blockIdx.x * blockDim.x + threadIdx.x;
  const size_t stride = (size_t)gridDim.x * blockDim.x;
  for (; i < n; i += stride) out[i] = (__bf16)in[i];
}

// Pack six [1024][64] f32 weights into one bf16 [1024][384]:
// cols 0:Qu 64:Ku 128:Vu 192:Qc 256:Kc 320:Vc
__global__ __launch_bounds__(256) void pack_weights_kernel(
    const float* __restrict__ wqu, const float* __restrict__ wku,
    const float* __restrict__ wvu, const float* __restrict__ wqc,
    const float* __restrict__ wkc, const float* __restrict__ wvc,
    __bf16* __restrict__ W) {
  const int idx = blockIdx.x * 256 + threadIdx.x;
  if (idx >= 1024 * 384) return;
  const int k = idx / 384;
  const int c = idx - k * 384;
  const int w = c >> 6;
  const int col = c & 63;
  const float* src = (w == 0) ? wqu : (w == 1) ? wku : (w == 2) ? wvu
                     : (w == 3) ? wqc : (w == 4) ? wkc : wvc;
  W[idx] = (__bf16)src[k * 64 + col];
}

// ---------------------------------------------------------------------------
// Generic batched bf16 WMMA GEMM, double-buffered async global->LDS staging.
//   C[M x N] (f32 and/or bf16 out) = alpha * A[M x K] * op(B)
//   BT=false: B stored [K][N]; BT=true: B stored [N][K] (multiply by B^T).
// Block tile: 128 x BN, BK = 32. 256 threads = 8 wave32s.
// ---------------------------------------------------------------------------
template <int BN, bool BT>
__global__ __launch_bounds__(256) void gemm_bf16_kernel(
    const __bf16* __restrict__ A, const __bf16* __restrict__ Bsrc,
    float* __restrict__ Cf, __bf16* __restrict__ Cb,
    int K, int lda, int ldb, int ldc,
    long strideA, long strideB, long strideC, float alpha) {
  constexpr int BM = 128;
  constexpr int NW_N = BN / 64;          // wave groups along N (2 or 1)
  constexpr int NW_M = 8 / NW_N;         // wave groups along M (4 or 8)
  constexpr int WMT = BM / (16 * NW_M);  // 16x16 m-tiles per wave (2 or 1)
  constexpr int LDA_S = 40;              // 32 + 8 pad (keeps 16B row alignment)
  constexpr int LDB_S = BT ? 40 : (BN + 8);
  constexpr int A_ELE = BM * LDA_S;
  constexpr int B_ELE = BT ? (BN * LDB_S) : (32 * LDB_S);

  __shared__ __bf16 As[2 * A_ELE];
  __shared__ __bf16 Bs[2 * B_ELE];

  const int tid = (int)threadIdx.x;
  const int m0 = blockIdx.x * BM;
  const int n0 = blockIdx.y * BN;
  const int bz = blockIdx.z;
  A    += (size_t)bz * (size_t)strideA;
  Bsrc += (size_t)bz * (size_t)strideB;

  const int wid = tid >> 5;
  const int wm = wid % NW_M;
  const int wn = wid / NW_M;

  v8f acc[WMT][4];
#pragma unroll
  for (int i = 0; i < WMT; ++i)
#pragma unroll
    for (int j = 0; j < 4; ++j) acc[i][j] = zero_v8f();

  // stage one 32-wide k-chunk into LDS buffer `st`
  auto stage_in = [&](int k0, int st) {
    __bf16* as = &As[st * A_ELE];
    for (int v = tid; v < BM * 4; v += 256) {
      const int r = v >> 2, c = (v & 3) * 8;
      async_b128(&as[r * LDA_S + c], &A[(size_t)(m0 + r) * lda + k0 + c]);
    }
    __bf16* bs = &Bs[st * B_ELE];
    if (BT) {
      for (int v = tid; v < BN * 4; v += 256) {
        const int r = v >> 2, c = (v & 3) * 8;
        async_b128(&bs[r * LDB_S + c], &Bsrc[(size_t)(n0 + r) * ldb + k0 + c]);
      }
    } else {
      constexpr int SEGS = BN / 8;
      for (int v = tid; v < 32 * SEGS; v += 256) {
        const int r = v / SEGS, c = (v % SEGS) * 8;
        async_b128(&bs[r * LDB_S + c], &Bsrc[(size_t)(k0 + r) * ldb + n0 + c]);
      }
    }
  };

  stage_in(0, 0);
  int st = 0;
  for (int k0 = 0; k0 < K; k0 += 32) {
    wait_async0();        // my chunk-k0 loads have landed in LDS
    __syncthreads();      // everyone's have; everyone done reading other buffer
    if (k0 + 32 < K) stage_in(k0 + 32, st ^ 1);  // prefetch next chunk

    const __bf16* as = &As[st * A_ELE];
    const __bf16* bs = &Bs[st * B_ELE];
    v16bf a[WMT];
#pragma unroll
    for (int i = 0; i < WMT; ++i)
      a[i] = load_A_frag(&as[(wm * 16 * WMT + i * 16) * LDA_S], LDA_S);
#pragma unroll
    for (int jt = 0; jt < 4; ++jt) {
      v16bf bf;
      if (BT) bf = load_B_frag_T(&bs[(wn * 64 + jt * 16) * LDB_S], LDB_S);
      else    bf = load_B_frag_N(&bs[wn * 64 + jt * 16], LDB_S);
#pragma unroll
      for (int i = 0; i < WMT; ++i) acc[i][jt] = WMMA_BF16(a[i], bf, acc[i][jt]);
    }
    st ^= 1;
  }

  // Epilogue (C/D layout: lane n<16 col n rows 0..7; lane>=16 rows 8..15)
  const int lane = tid & 31;
  const int nl = lane & 15;
  const int mo = (lane >> 4) * 8;
  const size_t cbase = (size_t)bz * (size_t)strideC;
#pragma unroll
  for (int i = 0; i < WMT; ++i) {
#pragma unroll
    for (int jt = 0; jt < 4; ++jt) {
      const int gm = m0 + wm * 16 * WMT + i * 16;
      const int gn = n0 + wn * 64 + jt * 16 + nl;
#pragma unroll
      for (int r = 0; r < 8; ++r) {
        const float v = acc[i][jt][r] * alpha;
        const size_t idx = cbase + (size_t)(gm + mo + r) * ldc + gn;
        if (Cf) Cf[idx] = v;
        if (Cb) Cb[idx] = (__bf16)v;
      }
    }
  }
}

// ---------------------------------------------------------------------------
// Fused score kernel: per (b, 128-i-tile, 128-j-tile) compute
//   term1 = causal(Qc . Vu^T) * scale            -> bf16
//   sig   = sigmoid(scale * Qu . Ku^T) where j>i -> bf16 (0 elsewhere)
//   sc    = scale * Qc . Kc^T (raw, mask later)  -> f32
// ---------------------------------------------------------------------------
__device__ __forceinline__ void issue_tile_128x64(__bf16* __restrict__ dst,
                                                  const __bf16* __restrict__ P,
                                                  int rowBase, int colOff, int tid) {
  for (int v = tid; v < 1024; v += 256) {
    const int r = v >> 3, c = (v & 7) * 8;
    async_b128(&dst[r * 72 + c],
               &P[(size_t)(rowBase + r) * 384 + colOff + c]);
  }
}

__device__ __forceinline__ void mm_64(const __bf16* __restrict__ sA,
                                      const __bf16* __restrict__ sB,
                                      int wm, int wn, v8f (&acc)[2][4]) {
#pragma unroll
  for (int i = 0; i < 2; ++i)
#pragma unroll
    for (int j = 0; j < 4; ++j) acc[i][j] = zero_v8f();
#pragma unroll
  for (int kk = 0; kk < 64; kk += 32) {
    v16bf a0 = load_A_frag(sA + (size_t)(wm * 32 + 0) * 72 + kk, 72);
    v16bf a1 = load_A_frag(sA + (size_t)(wm * 32 + 16) * 72 + kk, 72);
#pragma unroll
    for (int jt = 0; jt < 4; ++jt) {
      v16bf bf = load_B_frag_T(sB + (size_t)(wn * 64 + jt * 16) * 72 + kk, 72);
      acc[0][jt] = WMMA_BF16(a0, bf, acc[0][jt]);
      acc[1][jt] = WMMA_BF16(a1, bf, acc[1][jt]);
    }
  }
}

__global__ __launch_bounds__(256) void qkv_scores_kernel(
    const __bf16* __restrict__ P, __bf16* __restrict__ term1,
    __bf16* __restrict__ sig, float* __restrict__ sc, float scale) {
  __shared__ __bf16 sQu[128 * 72];
  __shared__ __bf16 sQc[128 * 72];
  __shared__ __bf16 sB[128 * 72];

  const int tid = (int)threadIdx.x;
  const int b = blockIdx.z;
  const int i0 = blockIdx.x * 128;
  const int j0 = blockIdx.y * 128;
  const int rowI = b * 2048 + i0;
  const int rowJ = b * 2048 + j0;

  issue_tile_128x64(sQu, P, rowI, 0, tid);
  issue_tile_128x64(sQc, P, rowI, 192, tid);
  issue_tile_128x64(sB, P, rowJ, 128, tid);  // Vu
  wait_async0();
  __syncthreads();

  const int wid = tid >> 5;
  const int wm = wid & 3;
  const int wn = wid >> 2;
  const int lane = tid & 31;
  const int nl = lane & 15;
  const int mo = (lane >> 4) * 8;
  const size_t rowb = (size_t)b * 2048;
  v8f acc[2][4];

  // --- term1 = causal(Qc . Vu^T) * scale ---
  mm_64(sQc, sB, wm, wn, acc);
#pragma unroll
  for (int i = 0; i < 2; ++i)
#pragma unroll
    for (int jt = 0; jt < 4; ++jt) {
      const int gmb = i0 + wm * 32 + i * 16 + mo;
      const int gj = j0 + wn * 64 + jt * 16 + nl;
#pragma unroll
      for (int r = 0; r < 8; ++r) {
        const int gi = gmb + r;
        float v = acc[i][jt][r] * scale;
        if (gj > gi) v = 0.f;
        term1[(rowb + gi) * 2048 + gj] = (__bf16)v;
      }
    }

  __syncthreads();
  issue_tile_128x64(sB, P, rowJ, 64, tid);  // Ku
  wait_async0();
  __syncthreads();

  // --- sig = sigmoid(scale * Qu . Ku^T) where j>i else 0 ---
  mm_64(sQu, sB, wm, wn, acc);
#pragma unroll
  for (int i = 0; i < 2; ++i)
#pragma unroll
    for (int jt = 0; jt < 4; ++jt) {
      const int gmb = i0 + wm * 32 + i * 16 + mo;
      const int gj = j0 + wn * 64 + jt * 16 + nl;
#pragma unroll
      for (int r = 0; r < 8; ++r) {
        const int gi = gmb + r;
        const float l = acc[i][jt][r] * scale;
        const float s = (gj > gi) ? (1.f / (1.f + __expf(-l))) : 0.f;
        sig[(rowb + gi) * 2048 + gj] = (__bf16)s;
      }
    }

  __syncthreads();
  issue_tile_128x64(sB, P, rowJ, 256, tid);  // Kc
  wait_async0();
  __syncthreads();

  // --- sc = scale * Qc . Kc^T (unmasked; softmax kernel applies causal) ---
  mm_64(sQc, sB, wm, wn, acc);
#pragma unroll
  for (int i = 0; i < 2; ++i)
#pragma unroll
    for (int jt = 0; jt < 4; ++jt) {
      const int gmb = i0 + wm * 32 + i * 16 + mo;
      const int gj = j0 + wn * 64 + jt * 16 + nl;
#pragma unroll
      for (int r = 0; r < 8; ++r) {
        const int gi = gmb + r;
        sc[(rowb + gi) * 2048 + gj] = acc[i][jt][r] * scale;
      }
    }
}

// ---------------------------------------------------------------------------
// Row softmax of (causal(sc) - silu(su)) -> bf16 probs. One block per row.
// ---------------------------------------------------------------------------
__global__ __launch_bounds__(256) void softmax_silu_kernel(
    const float* __restrict__ sc, const float* __restrict__ su,
    __bf16* __restrict__ probs) {
  const size_t row = blockIdx.x;
  const int i = (int)(row & 2047);
  const float* scr = sc + row * 2048;
  const float* sur = su + row * 2048;
  __bf16* pr = probs + row * 2048;
  __shared__ float red[256];

  float z[8];
  float mx = -3.4e38f;
#pragma unroll
  for (int t = 0; t < 8; ++t) {
    const int j = (int)threadIdx.x + t * 256;
    float v;
    if (j <= i) {
      const float u = sur[j];
      v = scr[j] - u / (1.f + __expf(-u));  // sc - silu(su)
    } else {
      v = -__builtin_inff();
    }
    z[t] = v;
    mx = fmaxf(mx, v);
  }
  red[threadIdx.x] = mx;
  __syncthreads();
  for (int s = 128; s > 0; s >>= 1) {
    if ((int)threadIdx.x < s)
      red[threadIdx.x] = fmaxf(red[threadIdx.x], red[threadIdx.x + s]);
    __syncthreads();
  }
  mx = red[0];
  __syncthreads();

  float sum = 0.f;
#pragma unroll
  for (int t = 0; t < 8; ++t) {
    const float e = __expf(z[t] - mx);
    z[t] = e;
    sum += e;
  }
  red[threadIdx.x] = sum;
  __syncthreads();
  for (int s = 128; s > 0; s >>= 1) {
    if ((int)threadIdx.x < s) red[threadIdx.x] += red[threadIdx.x + s];
    __syncthreads();
  }
  const float inv = 1.f / red[0];
#pragma unroll
  for (int t = 0; t < 8; ++t) {
    const int j = (int)threadIdx.x + t * 256;
    pr[j] = (__bf16)(z[t] * inv);
  }
}

// ---------------------------------------------------------------------------
// Host launch
// ---------------------------------------------------------------------------
extern "C" void kernel_launch(void* const* d_in, const int* in_sizes, int n_in,
                              void* d_out, int out_size, void* d_ws, size_t ws_size,
                              hipStream_t stream) {
  (void)in_sizes; (void)n_in; (void)out_size; (void)ws_size;
  constexpr int B = 4, S = 2048, D = 1024;
  constexpr float scale = 0.125f;  // DH^-0.5 = 64^-0.5
  const long SS = (long)S * S;

  const float* x   = (const float*)d_in[0];
  const float* wqu = (const float*)d_in[1];
  const float* wku = (const float*)d_in[2];
  const float* wvu = (const float*)d_in[3];
  const float* wqc = (const float*)d_in[4];
  const float* wkc = (const float*)d_in[5];
  const float* wvc = (const float*)d_in[6];
  float* out = (float*)d_out;

  char* ws = (char*)d_ws;
  size_t off = 0;
  auto take = [&](size_t bytes) {
    char* p = ws + off;
    off += (bytes + 255) & ~(size_t)255;
    return p;
  };
  __bf16* xbf   = (__bf16*)take((size_t)B * S * D * 2);   // 16.8 MB
  __bf16* Wbf   = (__bf16*)take((size_t)D * 384 * 2);     // 0.8 MB
  __bf16* Pbf   = (__bf16*)take((size_t)B * S * 384 * 2); // 6.3 MB
  __bf16* term1 = (__bf16*)take((size_t)B * SS * 2);      // 33.6 MB
  __bf16* sig   = (__bf16*)take((size_t)B * SS * 2);      // 33.6 MB
  float*  sc    = (float*) take((size_t)B * SS * 4);      // 67.1 MB
  float*  su    = (float*) take((size_t)B * SS * 4);      // 67.1 MB
  __bf16* probs = (__bf16*)take((size_t)B * SS * 2);      // 33.6 MB

  // 1. x -> bf16
  f32_to_bf16_kernel<<<dim3(4096), dim3(256), 0, stream>>>(
      x, xbf, (size_t)B * S * D);
  // 2. pack weights -> bf16 [1024][384]
  pack_weights_kernel<<<dim3((1024 * 384 + 255) / 256), dim3(256), 0, stream>>>(
      wqu, wku, wvu, wqc, wkc, wvc, Wbf);
  // 3. projections: P[8192][384] = xbf[8192][1024] @ Wbf[1024][384]
  gemm_bf16_kernel<128, false><<<dim3(64, 3, 1), dim3(256), 0, stream>>>(
      xbf, Wbf, nullptr, Pbf, /*K=*/1024, /*lda=*/1024, /*ldb=*/384, /*ldc=*/384,
      0, 0, 0, 1.f);
  // 4. fused scores: term1, sig, sc
  qkv_scores_kernel<<<dim3(16, 16, 4), dim3(256), 0, stream>>>(
      Pbf, term1, sig, sc, scale);
  // 5. S_u[b] = term1[b] @ sig[b]^T   (M=N=K=2048, batched over B)
  gemm_bf16_kernel<128, true><<<dim3(16, 16, 4), dim3(256), 0, stream>>>(
      term1, sig, su, nullptr, /*K=*/2048, /*lda=*/2048, /*ldb=*/2048,
      /*ldc=*/2048, SS, SS, SS, 1.f);
  // 6. probs = softmax(causal(sc) - silu(su))
  softmax_silu_kernel<<<dim3(B * S), dim3(256), 0, stream>>>(sc, su, probs);
  // 7. out[b] = probs[b] @ Vc[b]  (Vc = cols 320..383 of packed P)
  gemm_bf16_kernel<64, false><<<dim3(16, 1, 4), dim3(256), 0, stream>>>(
      probs, Pbf + 320, out, nullptr, /*K=*/2048, /*lda=*/2048, /*ldb=*/384,
      /*ldc=*/64, SS, (long)S * 384, (long)S * 64, 1.f);
}
